// BailingMoeBlock_67748814127136
// MI455X (gfx1250) — compile-verified, matrix operand order checked by reading, not run on hardware
//
#include <hip/hip_runtime.h>
#include <math.h>

// ---------------------------------------------------------------------------
// Types for CDNA5 WMMA (wave32):  D(16x16 f32) = A(16x32 bf16) x B(32x16 bf16) + C
// ---------------------------------------------------------------------------
typedef __bf16 bf16_t;
typedef __attribute__((ext_vector_type(16))) __bf16 bf16x16;
typedef __attribute__((ext_vector_type(8)))  __bf16 bf16x8;
typedef __attribute__((ext_vector_type(4)))  __bf16 bf16x4;
typedef __attribute__((ext_vector_type(8)))  float  f32x8;

#define S_LEN 2048
#define HDIM  2048
#define NHEAD 16
#define NKVH  4
#define HEADD 128
#define NEXP  16
#define TOPK  4
#define IDIM  1024
#define QKVN  3072   // (16 + 2*4) * 128

__device__ __forceinline__ f32x8 wmma_bf16(bf16x16 a, bf16x16 b, f32x8 c) {
  return __builtin_amdgcn_wmma_f32_16x16x32_bf16(false, a, false, b, (short)0, c,
                                                 false, false);
}

// Load one A/B fragment from LDS. Per the ISA 16-bit A layout (16x32):
// lane L: m = L%16, khalf = L/16; elements = row[khalf*8 .. +7] then
// row[16 + khalf*8 .. +7]   (two aligned 16-byte chunks).
__device__ __forceinline__ bf16x16 load_frag(const bf16_t* row, int khalf) {
  bf16x8 lo = *(const bf16x8*)(row + khalf * 8);
  bf16x8 hi = *(const bf16x8*)(row + 16 + khalf * 8);
  bf16x16 r;
#pragma unroll
  for (int i = 0; i < 8; ++i) { r[i] = lo[i]; r[i + 8] = hi[i]; }
  return r;
}

// Async 16-byte global(bf16) -> LDS copy (CDNA5 async datapath, ASYNCcnt).
__device__ __forceinline__ void async_copy_b128(bf16_t* lds_dst,
                                                const bf16_t* gsrc) {
  const unsigned lds = (unsigned)(size_t)lds_dst;
  const unsigned long long ga = (unsigned long long)gsrc;
  asm volatile("global_load_async_to_lds_b128 %0, %1, off"
               :: "v"(lds), "v"(ga) : "memory");
}
__device__ __forceinline__ void wait_async() {
  asm volatile("s_wait_asynccnt 0x0" ::: "memory");
}

// ---------------------------------------------------------------------------
// Generic tiled GEMM:  C[M x N](f32) = A[M x K](bf16, global) * B[K x N](f32, global)
// BM=64 BN=128 BK=32, 256 threads = 8 waves (2x4), each wave: 2x2 WMMA tiles.
// mode 0: plain rows.  mode 1: gather A rows from rowlist (compact output rows).
// mode 2: scatter output rows via rowlist, scale by combine weight.
// Optional Cin added (residual).  Early-exit on per-expert row count.
// A tile staged with global_load_async_to_lds_b128; B tile converted f32->bf16
// and stored transposed [n][k] with packed b64 stores.
// ---------------------------------------------------------------------------
__global__ __launch_bounds__(256) void gemm_bf16_kernel(
    const bf16_t* __restrict__ A, int lda, const float* __restrict__ B, int ldb,
    const float* __restrict__ Cin, float* __restrict__ Cout, int ldc, int M,
    int N, int K, int mode, const int* __restrict__ rowlist,
    const int* __restrict__ countptr, const float* __restrict__ scaleptr,
    int scalestride) {
  const int cnt = countptr ? *countptr : M;
  const int m0 = blockIdx.x * 64;
  if (m0 >= cnt) return;
  const int n0 = blockIdx.y * 128;

  __shared__ bf16_t sA[64 * 32];
  __shared__ bf16_t sB[128 * 32];

  const int tid = threadIdx.x;
  const int wave = tid >> 5, lane = tid & 31;
  const int wm = wave >> 2, wn = wave & 3;  // 2 x 4 wave grid
  const int l16 = lane & 15, khalf = lane >> 4;

  // Per-thread A staging coords (one async b128 per thread per K-step).
  const int a_row = tid >> 2, a_kg = (tid & 3) * 8;
  int a_r = m0 + a_row;
  if (a_r >= cnt) a_r = cnt - 1;  // clamp: these C rows are never stored
  const int a_grow = (mode == 1 && rowlist) ? rowlist[a_r] : a_r;
  const bf16_t* a_base = A + (size_t)a_grow * lda + a_kg;
  bf16_t* a_lds = sA + a_row * 32 + a_kg;

  // Per-thread B staging coords: 4 k-rows x 4 n-cols sub-tile.
  const int b_kq = (tid & 7) * 4;
  const int b_ng = (tid >> 3) * 4;

  f32x8 acc[2][2] = {};

  for (int k0 = 0; k0 < K; k0 += 32) {
    // A tile: async global->LDS (bf16, no conversion needed).
    async_copy_b128(a_lds, a_base + k0);

    // B tile: 32x128 f32 -> bf16, transposed store [n][k], packed b64.
    {
      const float* bp = B + (size_t)(k0 + b_kq) * ldb + n0 + b_ng;
      float4 c0 = *(const float4*)(bp);
      float4 c1 = *(const float4*)(bp + ldb);
      float4 c2 = *(const float4*)(bp + 2 * ldb);
      float4 c3 = *(const float4*)(bp + 3 * ldb);
      bf16x4 p0 = {(bf16_t)c0.x, (bf16_t)c1.x, (bf16_t)c2.x, (bf16_t)c3.x};
      bf16x4 p1 = {(bf16_t)c0.y, (bf16_t)c1.y, (bf16_t)c2.y, (bf16_t)c3.y};
      bf16x4 p2 = {(bf16_t)c0.z, (bf16_t)c1.z, (bf16_t)c2.z, (bf16_t)c3.z};
      bf16x4 p3 = {(bf16_t)c0.w, (bf16_t)c1.w, (bf16_t)c2.w, (bf16_t)c3.w};
      *(bf16x4*)(sB + (b_ng + 0) * 32 + b_kq) = p0;
      *(bf16x4*)(sB + (b_ng + 1) * 32 + b_kq) = p1;
      *(bf16x4*)(sB + (b_ng + 2) * 32 + b_kq) = p2;
      *(bf16x4*)(sB + (b_ng + 3) * 32 + b_kq) = p3;
    }
    wait_async();
    __syncthreads();

    bf16x16 af[2], bfm[2];
#pragma unroll
    for (int i = 0; i < 2; ++i)
      af[i] = load_frag(sA + (wm * 32 + i * 16 + l16) * 32, khalf);
#pragma unroll
    for (int j = 0; j < 2; ++j)
      bfm[j] = load_frag(sB + (wn * 32 + j * 16 + l16) * 32, khalf);
#pragma unroll
    for (int i = 0; i < 2; ++i)
#pragma unroll
      for (int j = 0; j < 2; ++j)
        acc[i][j] = wmma_bf16(af[i], bfm[j], acc[i][j]);
    __syncthreads();
  }

  // Epilogue. C layout: lane holds col n=l16, rows r8 + 8*khalf.
#pragma unroll
  for (int i = 0; i < 2; ++i) {
#pragma unroll
    for (int j = 0; j < 2; ++j) {
      const int col = n0 + wn * 32 + j * 16 + l16;
#pragma unroll
      for (int r8 = 0; r8 < 8; ++r8) {
        const int r = m0 + wm * 32 + i * 16 + r8 + 8 * khalf;
        if (r >= cnt) continue;
        const int orow = (mode == 2 && rowlist) ? rowlist[r] : r;
        float v = acc[i][j][r8];
        if (scaleptr) v *= scaleptr[(size_t)orow * scalestride];
        if (Cin) v += Cin[(size_t)orow * ldc + col];
        Cout[(size_t)orow * ldc + col] = v;
      }
    }
  }
}

// ---------------------------------------------------------------------------
// RMSNorm: one block per token, write bf16 normalized activations.
// ---------------------------------------------------------------------------
__global__ __launch_bounds__(256) void rmsnorm_kernel(
    const float* __restrict__ x, const float* __restrict__ w,
    bf16_t* __restrict__ out) {
  const int t = blockIdx.x;
  const float* xr = x + (size_t)t * HDIM;
  __shared__ float red[256];
  float s = 0.f;
  for (int i = threadIdx.x; i < HDIM; i += 256) { float v = xr[i]; s += v * v; }
  red[threadIdx.x] = s;
  __syncthreads();
  for (int o = 128; o > 0; o >>= 1) {
    if (threadIdx.x < o) red[threadIdx.x] += red[threadIdx.x + o];
    __syncthreads();
  }
  const float rs = rsqrtf(red[0] / (float)HDIM + 1e-6f);
  for (int i = threadIdx.x; i < HDIM; i += 256)
    out[(size_t)t * HDIM + i] = (bf16_t)(xr[i] * rs * w[i]);
}

// ---------------------------------------------------------------------------
// RoPE applied in-place to q (heads 0..15) and k (heads 16..19) of qkv rows.
// ---------------------------------------------------------------------------
__global__ __launch_bounds__(256) void rope_kernel(float* __restrict__ qkv) {
  const int t = blockIdx.x;
  float* row = qkv + (size_t)t * QKVN;
  const float lt = logf(10000.0f);
  for (int idx = threadIdx.x; idx < 20 * 64; idx += 256) {
    const int h = idx >> 6, d = idx & 63;
    const float inv = __expf(-(float)(2 * d) / 128.0f * lt);
    const float ang = (float)t * inv;
    const float c = __cosf(ang), s = __sinf(ang);
    float* base = row + h * 128;
    const float x1 = base[d], x2 = base[d + 64];
    base[d] = x1 * c - x2 * s;
    base[d + 64] = x1 * s + x2 * c;
  }
}

// ---------------------------------------------------------------------------
// Flash attention (causal, GQA 4:1). Block = 8 waves; wave owns a 16-row Q
// tile; block cooperatively stages 32-key K/V tiles in LDS. WMMA for QK^T
// and PV, online softmax in VALU.
// ---------------------------------------------------------------------------
__global__ __launch_bounds__(256) void attn_kernel(
    const float* __restrict__ qkv, bf16_t* __restrict__ ctx) {
  const int head = blockIdx.y;
  const int kvh = head >> 2;
  const int tid = threadIdx.x;
  const int wave = tid >> 5, lane = tid & 31;
  const int l16 = lane & 15, khalf = lane >> 4;
  const int q0 = blockIdx.x * 128 + wave * 16;

  __shared__ bf16_t sk[32 * 128];      // [key][d]
  __shared__ bf16_t sv[128 * 32];      // [d][key] (transposed)
  __shared__ bf16_t sp[8][16 * 32];    // per-wave P tile

  const float scale = 0.08838834764831845f;  // 1/sqrt(128)

  // Q fragments: 4 chunks of K=32 along HEADD, vectorized f32 loads.
  bf16x16 qf[4];
  {
    const int tq = q0 + l16;
    const float* qrow = qkv + (size_t)tq * QKVN + head * 128;
#pragma unroll
    for (int c = 0; c < 4; ++c) {
      const float* p0 = qrow + c * 32 + khalf * 8;
      const float* p1 = qrow + c * 32 + 16 + khalf * 8;
      float4 a0 = ((const float4*)p0)[0], a1 = ((const float4*)p0)[1];
      float4 a2 = ((const float4*)p1)[0], a3 = ((const float4*)p1)[1];
      bf16x16 f;
      f[0]  = (bf16_t)(a0.x * scale); f[1]  = (bf16_t)(a0.y * scale);
      f[2]  = (bf16_t)(a0.z * scale); f[3]  = (bf16_t)(a0.w * scale);
      f[4]  = (bf16_t)(a1.x * scale); f[5]  = (bf16_t)(a1.y * scale);
      f[6]  = (bf16_t)(a1.z * scale); f[7]  = (bf16_t)(a1.w * scale);
      f[8]  = (bf16_t)(a2.x * scale); f[9]  = (bf16_t)(a2.y * scale);
      f[10] = (bf16_t)(a2.z * scale); f[11] = (bf16_t)(a2.w * scale);
      f[12] = (bf16_t)(a3.x * scale); f[13] = (bf16_t)(a3.y * scale);
      f[14] = (bf16_t)(a3.z * scale); f[15] = (bf16_t)(a3.w * scale);
      qf[c] = f;
    }
  }

  f32x8 accO[8] = {};
  float mi[8], li[8];
#pragma unroll
  for (int r = 0; r < 8; ++r) { mi[r] = -1e30f; li[r] = 0.f; }

  const int ktiles = blockIdx.x * 4 + 4;  // causal: keys <= block's max q
  for (int kt = 0; kt < ktiles; ++kt) {
    // K tile: row-major, two packed 16B stores per thread.
    {
      const int krow = tid & 31, dg = (tid >> 5) * 16;
      const float* kp =
          qkv + (size_t)(kt * 32 + krow) * QKVN + 2048 + kvh * 128 + dg;
      float4 a0 = ((const float4*)kp)[0], a1 = ((const float4*)kp)[1];
      float4 a2 = ((const float4*)kp)[2], a3 = ((const float4*)kp)[3];
      bf16x8 h0 = {(bf16_t)a0.x, (bf16_t)a0.y, (bf16_t)a0.z, (bf16_t)a0.w,
                   (bf16_t)a1.x, (bf16_t)a1.y, (bf16_t)a1.z, (bf16_t)a1.w};
      bf16x8 h1 = {(bf16_t)a2.x, (bf16_t)a2.y, (bf16_t)a2.z, (bf16_t)a2.w,
                   (bf16_t)a3.x, (bf16_t)a3.y, (bf16_t)a3.z, (bf16_t)a3.w};
      *(bf16x8*)(sk + krow * 128 + dg) = h0;
      *(bf16x8*)(sk + krow * 128 + dg + 8) = h1;
    }
    // V tile transposed: 4 keys x 4 dims per thread, packed b64 stores.
    {
      const int kq = (tid & 7) * 4, dg = (tid >> 3) * 4;
      const float* vp =
          qkv + (size_t)(kt * 32 + kq) * QKVN + 2560 + kvh * 128 + dg;
      float4 r0 = *(const float4*)(vp);
      float4 r1 = *(const float4*)(vp + QKVN);
      float4 r2 = *(const float4*)(vp + 2 * QKVN);
      float4 r3 = *(const float4*)(vp + 3 * QKVN);
      bf16x4 p0 = {(bf16_t)r0.x, (bf16_t)r1.x, (bf16_t)r2.x, (bf16_t)r3.x};
      bf16x4 p1 = {(bf16_t)r0.y, (bf16_t)r1.y, (bf16_t)r2.y, (bf16_t)r3.y};
      bf16x4 p2 = {(bf16_t)r0.z, (bf16_t)r1.z, (bf16_t)r2.z, (bf16_t)r3.z};
      bf16x4 p3 = {(bf16_t)r0.w, (bf16_t)r1.w, (bf16_t)r2.w, (bf16_t)r3.w};
      *(bf16x4*)(sv + (dg + 0) * 32 + kq) = p0;
      *(bf16x4*)(sv + (dg + 1) * 32 + kq) = p1;
      *(bf16x4*)(sv + (dg + 2) * 32 + kq) = p2;
      *(bf16x4*)(sv + (dg + 3) * 32 + kq) = p3;
    }
    __syncthreads();

    if (kt * 32 <= q0 + 15) {
      // Scores: two 16x16 tiles (keys n and n+16), K accumulated over 4 chunks.
      f32x8 s0 = {}, s1 = {};
#pragma unroll
      for (int c = 0; c < 4; ++c) {
        bf16x16 b0 = load_frag(sk + l16 * 128 + c * 32, khalf);
        bf16x16 b1 = load_frag(sk + (16 + l16) * 128 + c * 32, khalf);
        s0 = wmma_bf16(qf[c], b0, s0);
        s1 = wmma_bf16(qf[c], b1, s1);
      }

      const int key0 = kt * 32 + l16;
#pragma unroll
      for (int r = 0; r < 8; ++r) {
        const int qrow = q0 + r + 8 * khalf;
        float v0 = s0[r], v1 = s1[r];
        if (key0 > qrow) v0 = -1e30f;
        if (key0 + 16 > qrow) v1 = -1e30f;
        float mr = fmaxf(v0, v1);
#pragma unroll
        for (int off = 1; off < 16; off <<= 1)
          mr = fmaxf(mr, __shfl_xor(mr, off, 32));
        const float mnew = fmaxf(mi[r], mr);
        const float p0 = __expf(v0 - mnew), p1 = __expf(v1 - mnew);
        float rs = p0 + p1;
#pragma unroll
        for (int off = 1; off < 16; off <<= 1) rs += __shfl_xor(rs, off, 32);
        const float alpha = __expf(mi[r] - mnew);
        li[r] = li[r] * alpha + rs;
        mi[r] = mnew;
#pragma unroll
        for (int dt = 0; dt < 8; ++dt) accO[dt][r] *= alpha;
        sp[wave][(r + 8 * khalf) * 32 + l16] = (bf16_t)p0;
        sp[wave][(r + 8 * khalf) * 32 + 16 + l16] = (bf16_t)p1;
      }

      // PV: P (16x32) x V (32 x 128) over 8 column tiles.
      const bf16x16 pa = load_frag(&sp[wave][l16 * 32], khalf);
#pragma unroll
      for (int dt = 0; dt < 8; ++dt) {
        bf16x16 bv = load_frag(sv + (dt * 16 + l16) * 32, khalf);
        accO[dt] = wmma_bf16(pa, bv, accO[dt]);
      }
    }
    __syncthreads();
  }

#pragma unroll
  for (int r = 0; r < 8; ++r) {
    const int tq = q0 + r + 8 * khalf;
    const float inv = 1.0f / li[r];
#pragma unroll
    for (int dt = 0; dt < 8; ++dt)
      ctx[(size_t)tq * HDIM + head * 128 + dt * 16 + l16] =
          (bf16_t)(accO[dt][r] * inv);
  }
}

// ---------------------------------------------------------------------------
// Router: logits -> softmax -> top-4 -> normalized combine weights + per-
// expert token lists (atomic append).
// ---------------------------------------------------------------------------
__global__ __launch_bounds__(256) void router_kernel(
    const bf16_t* __restrict__ xn, const float* __restrict__ w_gate,
    float* __restrict__ combine, int* __restrict__ counts,
    int* __restrict__ lists) {
  const int t = blockIdx.x;
  __shared__ float red[256];
  __shared__ float logits[NEXP];
  const int e = threadIdx.x & 15, chunk = threadIdx.x >> 4;
  const bf16_t* xr = xn + (size_t)t * HDIM;
  float s = 0.f;
  for (int i = 0; i < 128; ++i) {
    const int hcol = chunk * 128 + i;
    s += (float)xr[hcol] * w_gate[(size_t)hcol * NEXP + e];
  }
  red[threadIdx.x] = s;
  __syncthreads();
  for (int o = 8; o > 0; o >>= 1) {
    if (chunk < o) red[threadIdx.x] += red[threadIdx.x + o * 16];
    __syncthreads();
  }
  if (threadIdx.x < NEXP) logits[threadIdx.x] = red[threadIdx.x];
  __syncthreads();
  if (threadIdx.x == 0) {
    float mx = -1e30f;
    for (int i = 0; i < NEXP; ++i) mx = fmaxf(mx, logits[i]);
    float p[NEXP], sum = 0.f;
    for (int i = 0; i < NEXP; ++i) { p[i] = __expf(logits[i] - mx); sum += p[i]; }
    for (int i = 0; i < NEXP; ++i) p[i] /= sum;
    bool used[NEXP] = {};
    int sel[TOPK];
    float wsel[TOPK], tsum = 0.f;
    for (int kk = 0; kk < TOPK; ++kk) {
      int bi = -1; float bv = -1.f;
      for (int i = 0; i < NEXP; ++i)
        if (!used[i] && p[i] > bv) { bv = p[i]; bi = i; }
      used[bi] = true; sel[kk] = bi; wsel[kk] = bv; tsum += bv;
    }
    float cr[NEXP] = {};
    for (int kk = 0; kk < TOPK; ++kk) {
      cr[sel[kk]] = wsel[kk] / tsum;
      const int slot = atomicAdd(&counts[sel[kk]], 1);
      lists[sel[kk] * S_LEN + slot] = t;
    }
    for (int i = 0; i < NEXP; ++i) combine[(size_t)t * NEXP + i] = cr[i];
  }
}

// ---------------------------------------------------------------------------
// SwiGLU: h = silu(g) * u, f32 -> bf16, guarded by routed row count.
// ---------------------------------------------------------------------------
__global__ __launch_bounds__(256) void swiglu_kernel(
    const float* __restrict__ gu, bf16_t* __restrict__ h,
    const int* __restrict__ countptr, int Mmax) {
  const int r = blockIdx.x;
  const int cnt = countptr ? *countptr : Mmax;
  if (r >= cnt) return;
  const int i = blockIdx.y * 256 + threadIdx.x;  // 4*256 == IDIM
  const float g = gu[(size_t)r * (2 * IDIM) + i];
  const float u = gu[(size_t)r * (2 * IDIM) + IDIM + i];
  const float sg = g / (1.0f + __expf(-g));
  h[(size_t)r * IDIM + i] = (bf16_t)(sg * u);
}

__global__ void zero_counts_kernel(int* counts) {
  if (threadIdx.x < NEXP) counts[threadIdx.x] = 0;
}

// ---------------------------------------------------------------------------
// Host orchestration.
// ---------------------------------------------------------------------------
extern "C" void kernel_launch(void* const* d_in, const int* in_sizes, int n_in,
                              void* d_out, int out_size, void* d_ws,
                              size_t ws_size, hipStream_t stream) {
  (void)in_sizes; (void)n_in; (void)out_size; (void)ws_size;
  const float* hidden    = (const float*)d_in[0];
  // d_in[1] = position_ids (implicitly t = arange, used directly in rope)
  const float* ln1_w     = (const float*)d_in[2];
  const float* ln2_w     = (const float*)d_in[3];
  const float* w_qkv     = (const float*)d_in[4];
  const float* w_o       = (const float*)d_in[5];
  const float* w_gate    = (const float*)d_in[6];
  const float* w_gu      = (const float*)d_in[7];
  const float* w_down    = (const float*)d_in[8];
  const float* w_sh_gu   = (const float*)d_in[9];
  const float* w_sh_down = (const float*)d_in[10];
  float* out = (float*)d_out;

  char* ws = (char*)d_ws;
  size_t off = 0;
  auto take = [&](size_t bytes) {
    char* p = ws + off;
    off += (bytes + 255) & ~(size_t)255;
    return p;
  };
  bf16_t* xn      = (bf16_t*)take((size_t)S_LEN * HDIM * 2);
  float*  qkv     = (float*) take((size_t)S_LEN * QKVN * 4);
  bf16_t* ctx     = (bf16_t*)take((size_t)S_LEN * HDIM * 2);
  float*  resid   = (float*) take((size_t)S_LEN * HDIM * 4);
  float*  combine = (float*) take((size_t)S_LEN * NEXP * 4);
  int*    counts  = (int*)   take((size_t)NEXP * 4);
  int*    lists   = (int*)   take((size_t)NEXP * S_LEN * 4);
  float*  gu      = (float*) take((size_t)S_LEN * 2 * IDIM * 4);
  bf16_t* hbuf    = (bf16_t*)take((size_t)S_LEN * IDIM * 2);

  dim3 blk(256);

  zero_counts_kernel<<<1, 32, 0, stream>>>(counts);

  // ---- Attention half ----
  rmsnorm_kernel<<<S_LEN, blk, 0, stream>>>(hidden, ln1_w, xn);

  gemm_bf16_kernel<<<dim3(32, QKVN / 128), blk, 0, stream>>>(
      xn, HDIM, w_qkv, QKVN, nullptr, qkv, QKVN, S_LEN, QKVN, HDIM, 0, nullptr,
      nullptr, nullptr, 0);

  rope_kernel<<<S_LEN, blk, 0, stream>>>(qkv);

  attn_kernel<<<dim3(S_LEN / 128, NHEAD), blk, 0, stream>>>(qkv, ctx);

  // attn_out = ctx @ w_o ; resid = hidden + attn_out
  gemm_bf16_kernel<<<dim3(32, HDIM / 128), blk, 0, stream>>>(
      ctx, HDIM, w_o, HDIM, hidden, resid, HDIM, S_LEN, HDIM, HDIM, 0, nullptr,
      nullptr, nullptr, 0);

  // ---- MoE half ----
  rmsnorm_kernel<<<S_LEN, blk, 0, stream>>>(resid, ln2_w, xn);

  router_kernel<<<S_LEN, blk, 0, stream>>>(xn, w_gate, combine, counts, lists);

  for (int e = 0; e < NEXP; ++e) {
    const float* wg = w_gu + (size_t)e * HDIM * (2 * IDIM);
    const float* wd = w_down + (size_t)e * IDIM * HDIM;
    const int* lst = lists + e * S_LEN;
    const int* cnt = counts + e;
    // gu[r] = xn[lists[r]] @ w_gu[e]   (gathered A rows, compact output)
    gemm_bf16_kernel<<<dim3(32, (2 * IDIM) / 128), blk, 0, stream>>>(
        xn, HDIM, wg, 2 * IDIM, nullptr, gu, 2 * IDIM, S_LEN, 2 * IDIM, HDIM,
        1, lst, cnt, nullptr, 0);
    swiglu_kernel<<<dim3(S_LEN, IDIM / 256), blk, 0, stream>>>(gu, hbuf, cnt,
                                                               S_LEN);
    // resid[lists[r]] += combine[lists[r], e] * (h[r] @ w_down[e])
    gemm_bf16_kernel<<<dim3(32, HDIM / 128), blk, 0, stream>>>(
        hbuf, IDIM, wd, HDIM, resid, resid, HDIM, S_LEN, HDIM, IDIM, 2, lst,
        cnt, combine + e, NEXP);
  }

  // ---- Shared expert, fused final residual into d_out ----
  gemm_bf16_kernel<<<dim3(32, (2 * IDIM) / 128), blk, 0, stream>>>(
      xn, HDIM, w_sh_gu, 2 * IDIM, nullptr, gu, 2 * IDIM, S_LEN, 2 * IDIM,
      HDIM, 0, nullptr, nullptr, nullptr, 0);
  swiglu_kernel<<<dim3(S_LEN, IDIM / 256), blk, 0, stream>>>(gu, hbuf, nullptr,
                                                             S_LEN);
  // out = resid + h_sh @ w_sh_down
  gemm_bf16_kernel<<<dim3(32, HDIM / 128), blk, 0, stream>>>(
      hbuf, IDIM, w_sh_down, HDIM, resid, out, HDIM, S_LEN, HDIM, IDIM, 0,
      nullptr, nullptr, nullptr, 0);
}
